// Block_16174846837078
// MI455X (gfx1250) — matrix operand order repaired and checked by reading, not compiled, not run on hardware
//
#include <hip/hip_runtime.h>
#include <hip/hip_bf16.h>
#include <math.h>

// ---------------- types ----------------
typedef __bf16 bf16_t;
typedef __attribute__((ext_vector_type(8)))  __bf16 v8bf;
typedef __attribute__((ext_vector_type(16))) __bf16 v16bf;
typedef __attribute__((ext_vector_type(8)))  float  v8f;
typedef __attribute__((ext_vector_type(4)))  float  f32x4;

// ---------------- static shapes ----------------
#define BATCH   64
#define B2      32
#define NTOK    257
#define CDIM    768
#define NHEAD   12
#define HDIM    64
#define HIDDEN  3072
#define NPAD    288              // NTOK padded up so K is a multiple of 32
#define M2      (B2 * NTOK)      // 8224 tokens per half
#define MTOT    (BATCH * NTOK)   // 16448 tokens total

__device__ __forceinline__ float gelu_exact(float v) {
    return 0.5f * v * (1.0f + erff(v * 0.70710678118654752f));
}

// async global->LDS 16B copy (CDNA5, ASYNCcnt-tracked)
__device__ __forceinline__ void async_b128(unsigned lds_off, const void* gptr) {
    asm volatile("global_load_async_to_lds_b128 %0, %1, off"
                 :: "v"(lds_off), "v"(gptr) : "memory");
}
__device__ __forceinline__ void wait_asynccnt0() {
    asm volatile("s_wait_asynccnt 0x0" ::: "memory");
}

// =====================================================================
// V2: LDS-tiled, register-blocked GEMM for the large matmuls.
//   D = A(MxK) * Bt^T   (Bt stored N x K row-major, bf16)
// Block: 256 thr = 8 waves; block tile 128x128; wave tile 32(M)x64(N)
// => 2x4 f32 accumulators/wave. A/B k-chunks (128x32 bf16 each) staged in
// LDS via global_load_async_to_lds_b128, double-buffered, rows padded to
// 80B for conflict-free ds_load_b128 fragment reads.
// Requires: K % 32 == 0, N % 128 == 0 (all callers satisfy this).
// flags: bit0 = accumulate into outF, bit1 = exact GELU.
// =====================================================================
#define LDS_ROWB 80u            // bytes per LDS tile row (64 data + 16 pad)
#define LDS_ROWV 5u             // v8bf chunks per LDS tile row
#define LDS_TILEV 640u          // v8bf chunks per 128-row tile
#define LDS_PAIRV 1280u         // A tile + B tile

__global__ void __launch_bounds__(256)
gemm_wmma_bf16_lds(const bf16_t* __restrict__ A, int lda,
                   const bf16_t* __restrict__ Bt, int ldb,
                   float* outF, bf16_t* outB, int ldo,
                   const float* __restrict__ bias, const float* __restrict__ gw, int gidx,
                   int M, int N, int K, int flags)
{
    __shared__ v8bf smem_v[2 * LDS_PAIRV];           // 40 KB
    const unsigned smem_base = (unsigned)(uintptr_t)(void*)smem_v;

    const int tid  = threadIdx.x;
    const int wave = tid >> 5;
    const int lane = tid & 31;
    const int half = lane >> 4;                      // 0: lanes 0-15, 1: 16-31
    const int l15  = lane & 15;
    const int waveM = wave >> 1;                     // 0..3 -> 32-row slabs
    const int waveN = wave & 1;                      // 0..1 -> 64-col slabs

    const int M0 = blockIdx.x * 128;
    const int N0 = blockIdx.y * 128;

    // this thread's two 16B copy chunks per tile: row = tid>>1, cols 2*(tid&1), +1
    const int crow = tid >> 1;
    const int cc0  = (tid & 1) * 2;
    const int cc1  = cc0 + 1;
    int garow = M0 + crow; if (garow >= M) garow = M - 1;   // clamp M edge
    int gbrow = N0 + crow; if (gbrow >= N) gbrow = N - 1;
    const bf16_t* gA = A  + (long long)garow * lda;
    const bf16_t* gB = Bt + (long long)gbrow * ldb;
    const unsigned lrow = (unsigned)crow * LDS_ROWB;

    v8f acc[2][4];
    #pragma unroll
    for (int ms = 0; ms < 2; ++ms)
        #pragma unroll
        for (int ns = 0; ns < 4; ++ns)
            acc[ms][ns] = (v8f){0.f,0.f,0.f,0.f,0.f,0.f,0.f,0.f};

    auto issue_copy = [&](int kk, int d) {
        unsigned baseA = smem_base + (unsigned)d * (LDS_PAIRV * 16u);
        unsigned baseB = baseA + LDS_TILEV * 16u;
        async_b128(baseA + lrow + (unsigned)cc0 * 16u, gA + kk + cc0 * 8);
        async_b128(baseA + lrow + (unsigned)cc1 * 16u, gA + kk + cc1 * 8);
        async_b128(baseB + lrow + (unsigned)cc0 * 16u, gB + kk + cc0 * 8);
        async_b128(baseB + lrow + (unsigned)cc1 * 16u, gB + kk + cc1 * 8);
    };

    issue_copy(0, 0);
    wait_asynccnt0();
    __syncthreads();

    const int nk = K >> 5;
    for (int ik = 0; ik < nk; ++ik) {
        const int cur = ik & 1;
        if (ik + 1 < nk) issue_copy((ik + 1) << 5, cur ^ 1);   // overlap copy w/ WMMA

        const unsigned aBase = (unsigned)cur * LDS_PAIRV;
        const unsigned bBase = aBase + LDS_TILEV;

        // A fragments: lane<16 -> K {0..7,16..23}; lane>=16 -> {8..15,24..31}
        v16bf af[2];
        #pragma unroll
        for (int ms = 0; ms < 2; ++ms) {
            unsigned ra = (unsigned)(waveM * 32 + ms * 16 + l15);
            v8bf x0 = smem_v[aBase + ra * LDS_ROWV + (unsigned)half];
            v8bf x1 = smem_v[aBase + ra * LDS_ROWV + (unsigned)half + 2u];
            af[ms] = __builtin_shufflevector(x0, x1, 0,1,2,3,4,5,6,7,8,9,10,11,12,13,14,15);
        }
        // B fragments: lane<16 -> K {0..7,16..23}; lane>=16 -> {8..15,24..31}
        v16bf bfg[4];
        #pragma unroll
        for (int ns = 0; ns < 4; ++ns) {
            unsigned rb = (unsigned)(waveN * 64 + ns * 16 + l15);
            v8bf y0 = smem_v[bBase + rb * LDS_ROWV + (unsigned)half];
            v8bf y1 = smem_v[bBase + rb * LDS_ROWV + (unsigned)half + 2u];
            bfg[ns] = __builtin_shufflevector(y0, y1, 0,1,2,3,4,5,6,7,8,9,10,11,12,13,14,15);
        }
        #pragma unroll
        for (int ms = 0; ms < 2; ++ms)
            #pragma unroll
            for (int ns = 0; ns < 4; ++ns)
                acc[ms][ns] = __builtin_amdgcn_wmma_f32_16x16x32_bf16(
                    false, af[ms], false, bfg[ns], (short)0, acc[ms][ns], false, false);

        wait_asynccnt0();       // next buffer fully landed in LDS
        __syncthreads();        // visible to all waves; cur safe to overwrite next iter
    }

    const float g = (gw && gidx >= 0) ? gw[gidx] : 1.0f;
    #pragma unroll
    for (int ms = 0; ms < 2; ++ms) {
        #pragma unroll
        for (int ns = 0; ns < 4; ++ns) {
            int col = N0 + waveN * 64 + ns * 16 + l15;
            if (col < N) {
                float bv = bias ? bias[col] : 0.0f;
                #pragma unroll
                for (int r = 0; r < 8; ++r) {
                    int row = M0 + waveM * 32 + ms * 16 + r + 8 * half;
                    if (row < M) {
                        float v = acc[ms][ns][r] + bv;
                        if (flags & 2) v = gelu_exact(v);
                        v *= g;
                        long long idx = (long long)row * ldo + col;
                        if (outF) { if (flags & 1) outF[idx] += v; else outF[idx] = v; }
                        if (outB) outB[idx] = (bf16_t)v;
                    }
                }
            }
        }
    }
}

// =====================================================================
// V1: simple batched GEMM (one 16x16 tile / wave, direct global loads).
// Used for the small z-batched attention matmuls (scores: 257x257x64,
// AV: 257x64x288) where 128x128 tiling is wasteful.
// =====================================================================
__global__ void __launch_bounds__(256)
gemm_wmma_bf16(const bf16_t* __restrict__ A, int lda, long long sAb, long long sAh,
               const bf16_t* __restrict__ Bt, int ldb, long long sBb, long long sBh,
               float* outF, bf16_t* outB, int ldo, long long sOb, long long sOh,
               const float* __restrict__ bias, const float* __restrict__ gw, int gidx,
               int M, int N, int K, int batchH, int flags)
{
    int z  = blockIdx.z;
    int bz = z / batchH;
    int hz = z - bz * batchH;
    A  += (long long)bz * sAb + (long long)hz * sAh;
    Bt += (long long)bz * sBb + (long long)hz * sBh;
    long long oOff = (long long)bz * sOb + (long long)hz * sOh;

    int wave  = threadIdx.x >> 5;
    int lane  = threadIdx.x & 31;
    int ntile = blockIdx.y * 8 + wave;
    int N0    = ntile << 4;
    if (N0 >= N) return;                 // wave-uniform: before any WMMA
    int M0    = blockIdx.x << 4;

    int half = lane >> 4;
    int l15  = lane & 15;

    int arow = M0 + l15; if (arow >= M) arow = M - 1;
    int brow = N0 + l15; if (brow >= N) brow = N - 1;
    const bf16_t* Ar = A  + (long long)arow * lda;
    const bf16_t* Br = Bt + (long long)brow * ldb;

    v8f acc = {0.f, 0.f, 0.f, 0.f, 0.f, 0.f, 0.f, 0.f};

    for (int kk = 0; kk < K; kk += 32) {
        int ka = kk + half * 8;
        int kb = kk + half * 8;
        v8bf a0 = *(const v8bf*)(Ar + ka);
        v8bf a1 = *(const v8bf*)(Ar + ka + 16);
        v8bf b0 = *(const v8bf*)(Br + kb);
        v8bf b1 = *(const v8bf*)(Br + kb + 16);
        v16bf av = __builtin_shufflevector(a0, a1, 0,1,2,3,4,5,6,7,8,9,10,11,12,13,14,15);
        v16bf bv = __builtin_shufflevector(b0, b1, 0,1,2,3,4,5,6,7,8,9,10,11,12,13,14,15);
        acc = __builtin_amdgcn_wmma_f32_16x16x32_bf16(false, av, false, bv,
                                                      (short)0, acc, false, false);
    }

    float g   = (gw && gidx >= 0) ? gw[gidx] : 1.0f;
    int   col = N0 + l15;
    if (col < N) {
        float bv = bias ? bias[col] : 0.0f;
        #pragma unroll
        for (int r = 0; r < 8; ++r) {
            int row = M0 + r + half * 8;
            if (row < M) {
                float v = acc[r] + bv;
                if (flags & 2) v = gelu_exact(v);
                v *= g;
                long long idx = oOff + (long long)row * ldo + col;
                if (outF) { if (flags & 1) outF[idx] += v; else outF[idx] = v; }
                if (outB) outB[idx] = (bf16_t)v;
            }
        }
    }
}

// =====================================================================
// LayerNorm over C=768 per token -> bf16. First M2 rows (g1,b1), rest (g2,b2).
// =====================================================================
__global__ void __launch_bounds__(256)
ln_to_bf16(const float* __restrict__ x, bf16_t* __restrict__ xn,
           const float* __restrict__ g1, const float* __restrict__ b1,
           const float* __restrict__ g2, const float* __restrict__ b2)
{
    int row = blockIdx.x;
    const float* g = (row < M2) ? g1 : g2;
    const float* b = (row < M2) ? b1 : b2;
    int tid = threadIdx.x;
    const float* xr = x + (long long)row * CDIM;
    float v0 = xr[tid], v1 = xr[tid + 256], v2 = xr[tid + 512];

    __shared__ float s1[256], s2[256];
    s1[tid] = v0 + v1 + v2;
    s2[tid] = v0 * v0 + v1 * v1 + v2 * v2;
    __syncthreads();
    for (int off = 128; off > 0; off >>= 1) {
        if (tid < off) { s1[tid] += s1[tid + off]; s2[tid] += s2[tid + off]; }
        __syncthreads();
    }
    float mean = s1[0] * (1.0f / CDIM);
    float var  = s2[0] * (1.0f / CDIM) - mean * mean;
    float rstd = rsqrtf(var + 1e-5f);

    bf16_t* xnr = xn + (long long)row * CDIM;
    xnr[tid]       = (bf16_t)((v0 - mean) * rstd * g[tid]       + b[tid]);
    xnr[tid + 256] = (bf16_t)((v1 - mean) * rstd * g[tid + 256] + b[tid + 256]);
    xnr[tid + 512] = (bf16_t)((v2 - mean) * rstd * g[tid + 512] + b[tid + 512]);
}

// =====================================================================
// Row softmax over 257 valid cols of an NPAD-wide bf16 row; scale=1/8.
// Zero-fills cols 257..287 so the AV GEMM can run K=288.
// =====================================================================
__global__ void __launch_bounds__(256)
softmax_rows(bf16_t* __restrict__ P)
{
    long long row = blockIdx.x;
    bf16_t* pr = P + row * NPAD;
    int tid = threadIdx.x;
    const float scale = 0.125f;

    float a0 = (float)pr[tid] * scale;
    float a1 = (tid == 0) ? (float)pr[256] * scale : -INFINITY;

    __shared__ float sm[256];
    sm[tid] = fmaxf(a0, a1);
    __syncthreads();
    for (int off = 128; off > 0; off >>= 1) {
        if (tid < off) sm[tid] = fmaxf(sm[tid], sm[tid + off]);
        __syncthreads();
    }
    float mx = sm[0];
    __syncthreads();

    float e0 = expf(a0 - mx);
    float e1 = (tid == 0) ? expf(a1 - mx) : 0.0f;
    sm[tid] = e0 + e1;
    __syncthreads();
    for (int off = 128; off > 0; off >>= 1) {
        if (tid < off) sm[tid] += sm[tid + off];
        __syncthreads();
    }
    float inv = 1.0f / sm[0];

    pr[tid] = (bf16_t)(e0 * inv);
    if (tid == 0) pr[256] = (bf16_t)(e1 * inv);
    if (tid >= 1 && tid < 32) pr[256 + tid] = (bf16_t)0.0f;
}

// fp32 (K x N) -> bf16 transposed (N x K)
__global__ void __launch_bounds__(256)
transpose_to_bf16(const float* __restrict__ W, bf16_t* __restrict__ Wt, int K, int N)
{
    long long idx = (long long)blockIdx.x * blockDim.x + threadIdx.x;
    long long tot = (long long)K * N;
    if (idx >= tot) return;
    int k = (int)(idx / N);
    int n = (int)(idx - (long long)k * N);
    Wt[(long long)n * K + k] = (bf16_t)W[idx];
}

// build v^T (per (b,h): HDIM x NPAD, zero-padded) from qkv half-buffer
__global__ void __launch_bounds__(256)
build_vT(const bf16_t* __restrict__ qkvh, bf16_t* __restrict__ vT)
{
    long long idx = (long long)blockIdx.x * blockDim.x + threadIdx.x;
    long long tot = (long long)B2 * NHEAD * HDIM * NPAD;
    if (idx >= tot) return;
    int t   = (int)(idx % NPAD);
    long long r = idx / NPAD;
    int d   = (int)(r % HDIM);
    int z   = (int)(r / HDIM);
    int b   = z / NHEAD, h = z - b * NHEAD;
    bf16_t val = (bf16_t)0.0f;
    if (t < NTOK)
        val = qkvh[((long long)(b * NTOK + t)) * (3 * CDIM) + 2 * CDIM + h * HDIM + d];
    vT[idx] = val;
}

// vectorized fp32 copy (residual init: d_out = x)
__global__ void __launch_bounds__(256)
copy_f32x4(const f32x4* __restrict__ src, f32x4* __restrict__ dst, long long n4)
{
    long long i = (long long)blockIdx.x * blockDim.x + threadIdx.x;
    if (i < n4) dst[i] = src[i];
}

// =====================================================================
extern "C" void kernel_launch(void* const* d_in, const int* in_sizes, int n_in,
                              void* d_out, int out_size, void* d_ws, size_t ws_size,
                              hipStream_t stream)
{
    (void)in_sizes; (void)n_in; (void)out_size; (void)ws_size;

    const float* x      = (const float*)d_in[0];
    const float* gw     = (const float*)d_in[1];
    const float* n1l_g  = (const float*)d_in[2];
    const float* n1l_b  = (const float*)d_in[3];
    const float* n1s_g  = (const float*)d_in[4];
    const float* n1s_b  = (const float*)d_in[5];
    const float* w_qkv  = (const float*)d_in[6];
    const float* w_proj = (const float*)d_in[7];
    const float* b_proj = (const float*)d_in[8];
    const float* n2l_g  = (const float*)d_in[9];
    const float* n2l_b  = (const float*)d_in[10];
    const float* n2s_g  = (const float*)d_in[11];
    const float* n2s_b  = (const float*)d_in[12];
    const float* w_fc1  = (const float*)d_in[13];
    const float* b_fc1  = (const float*)d_in[14];
    const float* w_fc2  = (const float*)d_in[15];
    const float* b_fc2  = (const float*)d_in[16];
    float* out = (float*)d_out;

    // ---- workspace carve-up (all bf16, 256B aligned) ----
    char* ws = (char*)d_ws;
    size_t off = 0;
    auto alloc = [&](size_t elems) {
        char* p = ws + off;
        off += (elems * sizeof(bf16_t) + 255) & ~(size_t)255;
        return (bf16_t*)p;
    };
    bf16_t* wqkvT  = alloc((size_t)3 * CDIM * CDIM);          // 2304 x 768
    bf16_t* wprojT = alloc((size_t)CDIM * CDIM);              // 768 x 768
    bf16_t* wfc1T  = alloc((size_t)HIDDEN * CDIM);            // 3072 x 768
    bf16_t* wfc2T  = alloc((size_t)CDIM * HIDDEN);            // 768 x 3072
    bf16_t* xn     = alloc((size_t)MTOT * CDIM);              // LN out (reused LN1/LN2)
    bf16_t* qkv    = alloc((size_t)MTOT * 3 * CDIM);          // fused qkv
    bf16_t* Pbuf   = alloc((size_t)B2 * NHEAD * NTOK * NPAD); // probs, per half
    bf16_t* vTbuf  = alloc((size_t)B2 * NHEAD * HDIM * NPAD); // v^T, per half
    bf16_t* obuf   = alloc((size_t)MTOT * CDIM);              // head outputs
    bf16_t* hbuf   = alloc((size_t)M2 * HIDDEN);              // MLP hidden (reused x4)

    auto gemm1 = [&](const bf16_t* A, int lda, long long sAb, long long sAh,
                     const bf16_t* Bt, int ldb, long long sBb, long long sBh,
                     float* oF, bf16_t* oB, int ldo, long long sOb, long long sOh,
                     const float* bias, const float* gwp, int gidx,
                     int M, int N, int K, int batchH, int nbatch, int flags) {
        dim3 grid((M + 15) / 16, ((N + 15) / 16 + 7) / 8, nbatch);
        gemm_wmma_bf16<<<grid, 256, 0, stream>>>(A, lda, sAb, sAh, Bt, ldb, sBb, sBh,
                                                 oF, oB, ldo, sOb, sOh, bias, gwp, gidx,
                                                 M, N, K, batchH, flags);
    };
    auto gemm2 = [&](const bf16_t* A, int lda, const bf16_t* Bt, int ldb,
                     float* oF, bf16_t* oB, int ldo,
                     const float* bias, const float* gwp, int gidx,
                     int M, int N, int K, int flags) {
        dim3 grid((M + 127) / 128, (N + 127) / 128, 1);
        gemm_wmma_bf16_lds<<<grid, 256, 0, stream>>>(A, lda, Bt, ldb, oF, oB, ldo,
                                                     bias, gwp, gidx, M, N, K, flags);
    };

    const int keep[3] = {768, 384, 256};   // C / alpha_i  (== hd * H / alpha_i)

    // ---- one-time weight transposes to bf16 (N x K) ----
    {
        long long n;
        n = (long long)CDIM * 3 * CDIM;
        transpose_to_bf16<<<(unsigned)((n + 255) / 256), 256, 0, stream>>>(w_qkv,  wqkvT,  CDIM, 3 * CDIM);
        n = (long long)CDIM * CDIM;
        transpose_to_bf16<<<(unsigned)((n + 255) / 256), 256, 0, stream>>>(w_proj, wprojT, CDIM, CDIM);
        n = (long long)CDIM * HIDDEN;
        transpose_to_bf16<<<(unsigned)((n + 255) / 256), 256, 0, stream>>>(w_fc1,  wfc1T,  CDIM, HIDDEN);
        n = (long long)HIDDEN * CDIM;
        transpose_to_bf16<<<(unsigned)((n + 255) / 256), 256, 0, stream>>>(w_fc2,  wfc2T,  HIDDEN, CDIM);
    }

    // ---- residual init: out = x ----
    {
        long long n4 = (long long)MTOT * CDIM / 4;
        copy_f32x4<<<(unsigned)((n4 + 255) / 256), 256, 0, stream>>>((const f32x4*)x, (f32x4*)out, n4);
    }

    // ---- LN1 ----
    ln_to_bf16<<<MTOT, 256, 0, stream>>>(x, xn, n1l_g, n1l_b, n1s_g, n1s_b);

    // ---- fused qkv GEMM over both halves (LDS-tiled) ----
    gemm2(xn, CDIM, wqkvT, CDIM, nullptr, qkv, 3 * CDIM,
          nullptr, nullptr, -1, MTOT, 3 * CDIM, CDIM, 0);

    // ---- attention core, per half, z-batched over (b, h) ----
    for (int hf = 0; hf < 2; ++hf) {
        const bf16_t* qkvh = qkv  + (long long)hf * M2 * 3 * CDIM;
        bf16_t*       oh   = obuf + (long long)hf * M2 * CDIM;

        // scores = q @ k^T : A = q (257x64), Bt = k (257x64), views into qkv
        gemm1(qkvh,        3 * CDIM, (long long)NTOK * 3 * CDIM, HDIM,
              qkvh + CDIM, 3 * CDIM, (long long)NTOK * 3 * CDIM, HDIM,
              nullptr, Pbuf, NPAD, (long long)NHEAD * NTOK * NPAD, (long long)NTOK * NPAD,
              nullptr, nullptr, -1, NTOK, NTOK, HDIM, NHEAD, B2 * NHEAD, 0);

        softmax_rows<<<B2 * NHEAD * NTOK, 256, 0, stream>>>(Pbuf);

        { // v^T with zero K-padding
            long long n = (long long)B2 * NHEAD * HDIM * NPAD;
            build_vT<<<(unsigned)((n + 255) / 256), 256, 0, stream>>>(qkvh, vTbuf);
        }

        // o = P @ v : A = P (257x288), Bt = v^T (64x288)
        gemm1(Pbuf,  NPAD, (long long)NHEAD * NTOK * NPAD, (long long)NTOK * NPAD,
              vTbuf, NPAD, (long long)NHEAD * HDIM * NPAD, (long long)HDIM * NPAD,
              nullptr, oh, CDIM, (long long)NTOK * CDIM, HDIM,
              nullptr, nullptr, -1, NTOK, HDIM, NPAD, NHEAD, B2 * NHEAD, 0);
    }

    // ---- projection: large (full), small (3 gumbel-weighted prefix-masked) ----
    gemm2(obuf, CDIM, wprojT, CDIM, out, nullptr, CDIM,
          b_proj, nullptr, -1, M2, CDIM, CDIM, 1 /*accumulate*/);
    for (int i = 0; i < 3; ++i) {
        int Kc = keep[i];  // head-mask => truncate K; ch-mask => truncate N
        gemm2(obuf + (long long)M2 * CDIM, CDIM, wprojT, CDIM,
              out + (long long)M2 * CDIM, nullptr, CDIM,
              b_proj, gw, i, M2, Kc, Kc, 1);
    }

    // ---- LN2 on residual-updated activations ----
    ln_to_bf16<<<MTOT, 256, 0, stream>>>(out, xn, n2l_g, n2l_b, n2s_g, n2s_b);

    // ---- MLP: large ----
    gemm2(xn, CDIM, wfc1T, CDIM, nullptr, hbuf, HIDDEN,
          b_fc1, nullptr, -1, M2, HIDDEN, CDIM, 2 /*gelu*/);
    gemm2(hbuf, HIDDEN, wfc2T, HIDDEN, out, nullptr, CDIM,
          b_fc2, nullptr, -1, M2, CDIM, HIDDEN, 1);

    // ---- MLP: small, per alpha ----
    for (int i = 0; i < 3; ++i) {
        int Kc = keep[i];
        gemm2(xn + (long long)M2 * CDIM, CDIM, wfc1T, CDIM, nullptr, hbuf, HIDDEN,
              b_fc1, nullptr, -1, M2, HIDDEN, Kc, 2);
        gemm2(hbuf, HIDDEN, wfc2T, HIDDEN, out + (long long)M2 * CDIM, nullptr, CDIM,
              b_fc2, gw, i, M2, Kc, HIDDEN, 1);
    }
}
